// RPN_75531294867651
// MI455X (gfx1250) — matrix operand order repaired and checked by reading, not compiled, not run on hardware
//
#include <hip/hip_runtime.h>
#include <hip/hip_bf16.h>
#include <math.h>

// ---------------------------------------------------------------------------
// RPN head for MI455X (gfx1250, wave32, WMMA).
// conv3x3 (574 GFLOP) runs as a barrier-free bf16 WMMA implicit GEMM over a
// zero-padded NHWC bf16 activation buffer (L2-resident); the two 1x1 convs
// are plain WMMA GEMMs. Decode / top-k / NMS are bandwidth-trivial tails.
// ---------------------------------------------------------------------------

typedef __attribute__((ext_vector_type(16))) __bf16 v16bf;
typedef __attribute__((ext_vector_type(8)))  __bf16 v8bf;
typedef __attribute__((ext_vector_type(8)))  float  v8f;

#define BB    8
#define CIN   2048
#define Hf    50
#define Wf    76
#define Hp    52            // padded
#define Wp    78            // padded
#define MIDc  512
#define Acnt  15
#define PIX   (Hf * Wf)     // 3800
#define MTOT  (BB * PIX)    // 30400 (divisible by 16)
#define K3    (CIN * 9)     // 18432
#define NANCH 57000         // PIX * Acnt
#define TOPK  2000
#define KEEP  1000

// d_out float layout: scores[8*57000] | deltas[8*57000*4] | proposals[8*1000*4]
#define OUT_SCORES 0
#define OUT_DELTAS 456000
#define OUT_PROPS  2280000

// ws byte offsets
#define OFF_B1    ((size_t)0)            // bf16 [512][18432]         18,874,368 B
#define OFF_BCLS  ((size_t)18874368)     // bf16 [16][512]
#define OFF_BREG  ((size_t)18890752)     // bf16 [64][512]
#define OFF_XIN   ((size_t)18956288)     // bf16 [8][52][78][2048]   132,907,008 B
#define OFF_XMID  ((size_t)151863296)    // bf16 [30400][512]         31,129,600 B
#define OFF_BOXES ((size_t)182992896)    // f32  [8][57000][4]
#define OFF_SSC   ((size_t)190288896)    // f32  [8][57000]
#define OFF_TOPB  ((size_t)192112896)    // f32  [8][2000][4]

#define XIN_WORDS ((BB * Hp * Wp * CIN * 2) / 4)   // u32 words for zero fill

static __device__ inline v16bf cat16(v8bf lo, v8bf hi) {
  return __builtin_shufflevector(lo, hi, 0, 1, 2, 3, 4, 5, 6, 7,
                                 8, 9, 10, 11, 12, 13, 14, 15);
}

// ---------------------------------------------------------------------------
// Zero-fill the padded NHWC activation buffer (u32 stores).
// ---------------------------------------------------------------------------
__global__ void fill_xin_kernel(unsigned* __restrict__ p) {
  int idx = blockIdx.x * blockDim.x + threadIdx.x;
  if (idx < XIN_WORDS) p[idx] = 0u;
}

// ---------------------------------------------------------------------------
// NCHW f32 -> padded NHWC bf16 via LDS 32x32 transpose tiles.
// Reads coalesced along W, writes coalesced along C.
// Block: 256 thr; tile = 32 channels x 32 x-positions at fixed (b, y).
// Grid: 8 * 50 * 3(xblocks) * 64(cblocks) = 76800 blocks.
// ---------------------------------------------------------------------------
__global__ __launch_bounds__(256) void pack_feat_kernel(
    const float* __restrict__ feat, __bf16* __restrict__ xin) {
  __shared__ float tile[32][33];
  unsigned t = blockIdx.x;
  const int cb = t & 63;  t >>= 6;
  const int xb = t % 3;   t /= 3;
  const int y  = t % 50;
  const int b  = t / 50;
  const int c0 = cb * 32;
  const int x0 = xb * 32;
  const int tid = threadIdx.x;

#pragma unroll
  for (int i = 0; i < 4; ++i) {
    int e = tid + i * 256;
    int cc = e >> 5;
    int xx = e & 31;
    int x = x0 + xx;
    float v = 0.0f;
    if (x < Wf) v = feat[((size_t)(b * CIN + c0 + cc) * Hf + y) * Wf + x];
    tile[cc][xx] = v;
  }
  __syncthreads();
#pragma unroll
  for (int i = 0; i < 4; ++i) {
    int e = tid + i * 256;
    int px = e >> 5;
    int cc = e & 31;
    int x = x0 + px;
    if (x < Wf)
      xin[((size_t)(b * Hp + (y + 1)) * Wp + (x + 1)) * CIN + c0 + cc] =
          (__bf16)tile[cc][px];
  }
}

// ---------------------------------------------------------------------------
// Weight packing: conv1_w [512][2048][3][3] f32 -> B1 bf16 [n][k], k = j*2048+c
// ---------------------------------------------------------------------------
__global__ void pack3_kernel(const float* __restrict__ w, __bf16* __restrict__ out) {
  int idx = blockIdx.x * blockDim.x + threadIdx.x;
  if (idx >= MIDc * K3) return;
  int n = idx / K3;
  int k = idx - n * K3;
  int j = k / CIN;
  int c = k - j * CIN;
  out[idx] = (__bf16)w[((size_t)n * CIN + c) * 9 + j];
}

// 1x1 weights [Nvalid][512] f32 -> bf16 [Npad][512], zero padded rows
__global__ void pack1x1_kernel(const float* __restrict__ w, __bf16* __restrict__ out,
                               int Nvalid, int Npad) {
  int idx = blockIdx.x * blockDim.x + threadIdx.x;
  if (idx >= Npad * MIDc) return;
  int n = idx / MIDc;
  out[idx] = (n < Nvalid) ? (__bf16)w[idx] : (__bf16)0.0f;
}

// ---------------------------------------------------------------------------
// conv3x3 implicit GEMM: M=30400, N=512, K=18432.
// Block = 256 thr (8 waves); wave w owns 4 N-tiles (w*16 + t*128), so one
// block covers all 512 channels of one 16-row M tile.  No LDS, no barriers:
// A fragments are unit-stride reads from the padded NHWC bf16 buffer.
// Grid = (1900, 1).
// ---------------------------------------------------------------------------
__global__ __launch_bounds__(256) void conv3x3_wmma(
    const __bf16* __restrict__ xin, const __bf16* __restrict__ B1,
    const float* __restrict__ bias, __bf16* __restrict__ xmid) {
  const int tid  = threadIdx.x;
  const int lane = tid & 31;
  const int wave = tid >> 5;
  const int m0 = blockIdx.x * 16;
  const int row = lane & 15;
  const int hh  = lane >> 4;  // 0 or 1

  // per-lane M-row coordinates
  const int m = m0 + row;
  const int b = m / PIX;
  const int p = m - b * PIX;
  const int y = p / Wf;
  const int x = p - y * Wf;

  // per-lane B row pointers for the 4 N-tiles
  const __bf16* bbase0 = B1 + (size_t)(wave * 16 + 0 * 128 + row) * K3 + hh * 16;
  const __bf16* bbase1 = B1 + (size_t)(wave * 16 + 1 * 128 + row) * K3 + hh * 16;
  const __bf16* bbase2 = B1 + (size_t)(wave * 16 + 2 * 128 + row) * K3 + hh * 16;
  const __bf16* bbase3 = B1 + (size_t)(wave * 16 + 3 * 128 + row) * K3 + hh * 16;

  v8f acc0 = {}, acc1 = {}, acc2 = {}, acc3 = {};

  for (int j = 0; j < 9; ++j) {
    const int ky = j / 3;   // 0..2 (padded coords)
    const int kx = j % 3;
    const __bf16* arow =
        xin + ((size_t)(b * Hp + y + ky) * Wp + (x + kx)) * CIN + hh * 16;
    const __bf16* b0 = bbase0 + j * CIN;
    const __bf16* b1 = bbase1 + j * CIN;
    const __bf16* b2 = bbase2 + j * CIN;
    const __bf16* b3 = bbase3 + j * CIN;
#pragma unroll 2
    for (int c0 = 0; c0 < CIN; c0 += 32) {
      v16bf af = cat16(*(const v8bf*)(arow + c0), *(const v8bf*)(arow + c0 + 8));
      v16bf bf0 = cat16(*(const v8bf*)(b0 + c0), *(const v8bf*)(b0 + c0 + 8));
      v16bf bf1 = cat16(*(const v8bf*)(b1 + c0), *(const v8bf*)(b1 + c0 + 8));
      v16bf bf2 = cat16(*(const v8bf*)(b2 + c0), *(const v8bf*)(b2 + c0 + 8));
      v16bf bf3 = cat16(*(const v8bf*)(b3 + c0), *(const v8bf*)(b3 + c0 + 8));
      acc0 = __builtin_amdgcn_wmma_f32_16x16x32_bf16(false, af, false, bf0,
                                                     (short)0, acc0, false, false);
      acc1 = __builtin_amdgcn_wmma_f32_16x16x32_bf16(false, af, false, bf1,
                                                     (short)0, acc1, false, false);
      acc2 = __builtin_amdgcn_wmma_f32_16x16x32_bf16(false, af, false, bf2,
                                                     (short)0, acc2, false, false);
      acc3 = __builtin_amdgcn_wmma_f32_16x16x32_bf16(false, af, false, bf3,
                                                     (short)0, acc3, false, false);
    }
  }

  // bias + ReLU, store mid activation as bf16 [m][512]
#pragma unroll
  for (int t = 0; t < 4; ++t) {
    v8f acc = (t == 0) ? acc0 : (t == 1) ? acc1 : (t == 2) ? acc2 : acc3;
    const int n = wave * 16 + t * 128 + row;
    const float bv = bias[n];
#pragma unroll
    for (int r = 0; r < 8; ++r) {
      int mm = m0 + r + hh * 8;
      float v = acc[r] + bv;
      v = v > 0.0f ? v : 0.0f;
      xmid[(size_t)mm * MIDc + n] = (__bf16)v;
    }
  }
}

// ---------------------------------------------------------------------------
// 1x1 conv GEMM: A = xmid bf16 [30400][512], B = packed [Npad][512].
// 4 waves/block, each wave one 16x16 tile; stores f32 (+bias) with layout
// out[b*3800*NV + pix*NV + n] for n < NV (matches d_out scores/deltas).
// ---------------------------------------------------------------------------
__global__ __launch_bounds__(128) void conv1x1_wmma(
    const __bf16* __restrict__ xmid, const __bf16* __restrict__ Bw,
    const float* __restrict__ bias, float* __restrict__ out, int NV) {
  const int lane = threadIdx.x & 31;
  const int wave = threadIdx.x >> 5;
  const int m0 = (blockIdx.x * 4 + wave) * 16;
  const int n0 = blockIdx.y * 16;
  const int row = lane & 15;
  const int hh = lane >> 4;

  v8f acc = {};
  const __bf16* arow = xmid + (size_t)(m0 + row) * MIDc + hh * 16;
  const __bf16* brow = Bw + (size_t)(n0 + row) * MIDc + hh * 16;
#pragma unroll 4
  for (int k0 = 0; k0 < MIDc; k0 += 32) {
    v16bf afrag = cat16(*(const v8bf*)(arow + k0), *(const v8bf*)(arow + k0 + 8));
    v16bf bfrag = cat16(*(const v8bf*)(brow + k0), *(const v8bf*)(brow + k0 + 8));
    acc = __builtin_amdgcn_wmma_f32_16x16x32_bf16(
        false, afrag, false, bfrag, (short)0, acc, false, false);
  }
  const int n = n0 + row;
  if (n < NV) {
    const float bv = bias[n];
#pragma unroll
    for (int r = 0; r < 8; ++r) {
      int m = m0 + r + hh * 8;
      int b = m / PIX;
      int p = m - b * PIX;
      out[(size_t)b * PIX * NV + (size_t)p * NV + n] = acc[r] + bv;
    }
  }
}

// ---------------------------------------------------------------------------
// Anchor decode + sigmoid.  One thread per (b, anchor).
// ---------------------------------------------------------------------------
__global__ void decode_kernel(const float* __restrict__ sc_in,
                              const float* __restrict__ dl_in,
                              const int* __restrict__ ih, const int* __restrict__ iw,
                              float* __restrict__ boxes, float* __restrict__ sscores) {
  int gi = blockIdx.x * blockDim.x + threadIdx.x;
  if (gi >= BB * NANCH) return;
  int b = gi / NANCH;
  int nn = gi - b * NANCH;
  int p = nn / Acnt;
  int a = nn - p * Acnt;
  int y = p / Wf;
  int x = p - y * Wf;
  int si = a / 3, ri = a - si * 3;
  float scale = 32.0f * (float)(1 << si);
  float sq = (ri == 0) ? 0.70710678118654752f : (ri == 1 ? 1.0f : 1.41421356237309505f);
  float aw = scale / sq;
  float ah = scale * sq;
  float acx = (x + 0.5f) * 16.0f;
  float acy = (y + 0.5f) * 16.0f;
  const float* d = dl_in + (size_t)gi * 4;
  float pcx = d[0] * aw + acx;
  float pcy = d[1] * ah + acy;
  float pw = expf(d[2]) * aw;
  float ph = expf(d[3]) * ah;
  float Wi = (float)(*iw);
  float Hi = (float)(*ih);
  float x1 = fminf(fmaxf(pcx - 0.5f * pw, 0.0f), Wi);
  float y1 = fminf(fmaxf(pcy - 0.5f * ph, 0.0f), Hi);
  float x2 = fminf(fmaxf(pcx + 0.5f * pw, 0.0f), Wi);
  float y2 = fminf(fmaxf(pcy + 0.5f * ph, 0.0f), Hi);
  float* bo = boxes + (size_t)gi * 4;
  bo[0] = x1; bo[1] = y1; bo[2] = x2; bo[3] = y2;
  float s = sc_in[gi];
  sscores[gi] = 1.0f / (1.0f + expf(-s));
}

// ---------------------------------------------------------------------------
// Per-image top-2000: 4-pass radix select on positive-float bits, LDS
// compaction, 2048-wide bitonic sort (descending), gather boxes.
// ---------------------------------------------------------------------------
__global__ __launch_bounds__(256) void topk_kernel(const float* __restrict__ sscores,
                                                   const float* __restrict__ boxes,
                                                   float* __restrict__ topb) {
  const int b = blockIdx.x;
  const int tid = threadIdx.x;
  const int nt = 256;
  const float* sc = sscores + (size_t)b * NANCH;

  __shared__ unsigned hist[256];
  __shared__ unsigned sPrefix, sRemain;
  __shared__ unsigned selU[2048];
  __shared__ int selI[2048];
  __shared__ unsigned cntGt, cntEq;

  if (tid == 0) { sPrefix = 0u; sRemain = TOPK; }
  for (int pass = 0; pass < 4; ++pass) {
    const int shift = 24 - pass * 8;
    for (int d2 = tid; d2 < 256; d2 += nt) hist[d2] = 0u;
    __syncthreads();
    const unsigned pref = sPrefix;
    const unsigned mask = (pass == 0) ? 0u : (0xFFFFFFFFu << (shift + 8));
    for (int i = tid; i < NANCH; i += nt) {
      unsigned u = __float_as_uint(sc[i]);  // scores in (0,1): bits order-preserving
      if ((u & mask) == pref) atomicAdd(&hist[(u >> shift) & 255], 1u);
    }
    __syncthreads();
    if (tid == 0) {
      unsigned rem = sRemain, cum = 0;
      int chosen = 0;
      for (int d2 = 255; d2 >= 0; --d2) {
        unsigned h = hist[d2];
        if (cum + h >= rem) { chosen = d2; break; }
        cum += h;
      }
      sRemain = rem - cum;
      sPrefix = pref | ((unsigned)chosen << shift);
    }
    __syncthreads();
  }
  const unsigned T = sPrefix;
  const unsigned G = TOPK - sRemain;  // count of scores strictly above threshold

  if (tid == 0) { cntGt = 0u; cntEq = 0u; }
  for (int i = tid; i < 2048; i += nt) { selU[i] = 0u; selI[i] = 0; }
  __syncthreads();
  for (int i = tid; i < NANCH; i += nt) {
    unsigned u = __float_as_uint(sc[i]);
    if (u > T) {
      unsigned p = atomicAdd(&cntGt, 1u);  // p < G < TOPK by construction
      selU[p] = u; selI[p] = i;
    } else if (u == T) {
      unsigned p = G + atomicAdd(&cntEq, 1u);
      if (p < TOPK) { selU[p] = u; selI[p] = i; }
    }
  }
  __syncthreads();

  // bitonic sort, descending by key
  for (unsigned k = 2; k <= 2048; k <<= 1) {
    for (unsigned j = k >> 1; j > 0; j >>= 1) {
      for (unsigned i = tid; i < 2048; i += nt) {
        unsigned ixj = i ^ j;
        if (ixj > i) {
          unsigned a0 = selU[i], a1 = selU[ixj];
          bool descBlock = ((i & k) == 0);
          bool doSwap = descBlock ? (a0 < a1) : (a0 > a1);
          if (doSwap) {
            selU[i] = a1; selU[ixj] = a0;
            int t0 = selI[i]; selI[i] = selI[ixj]; selI[ixj] = t0;
          }
        }
      }
      __syncthreads();
    }
  }

  float* tb = topb + (size_t)b * TOPK * 4;
  const float* bx = boxes + (size_t)b * NANCH * 4;
  for (int r = tid; r < TOPK; r += nt) {
    int idx = selI[r];
    tb[r * 4 + 0] = bx[idx * 4 + 0];
    tb[r * 4 + 1] = bx[idx * 4 + 1];
    tb[r * 4 + 2] = bx[idx * 4 + 2];
    tb[r * 4 + 3] = bx[idx * 4 + 3];
  }
}

// ---------------------------------------------------------------------------
// Per-image greedy NMS (IoU 0.6) + ordered compaction to 1000 slots (kept
// boxes first in score order; remaining slots zeroed, matching reference).
// ---------------------------------------------------------------------------
__global__ __launch_bounds__(256) void nms_kernel(const float* __restrict__ topb,
                                                  float* __restrict__ outp) {
  const int b = blockIdx.x;
  const int tid = threadIdx.x;
  const int nt = 256;
  __shared__ float bx[TOPK * 4];
  __shared__ float area[TOPK];
  __shared__ unsigned char supp[TOPK];
  __shared__ int pos[TOPK];
  __shared__ int srank;

  const float* tb = topb + (size_t)b * TOPK * 4;
  for (int i = tid; i < TOPK * 4; i += nt) bx[i] = tb[i];
  __syncthreads();
  for (int i = tid; i < TOPK; i += nt) {
    area[i] = (bx[i * 4 + 2] - bx[i * 4 + 0]) * (bx[i * 4 + 3] - bx[i * 4 + 1]);
    supp[i] = 0;
  }
  __syncthreads();

  for (int i = 0; i < TOPK; ++i) {
    if (!supp[i]) {
      float x1 = bx[i * 4 + 0], y1 = bx[i * 4 + 1];
      float x2 = bx[i * 4 + 2], y2 = bx[i * 4 + 3];
      float ai = area[i];
      for (int jj = i + 1 + tid; jj < TOPK; jj += nt) {
        if (!supp[jj]) {
          float xx1 = fmaxf(x1, bx[jj * 4 + 0]);
          float yy1 = fmaxf(y1, bx[jj * 4 + 1]);
          float xx2 = fminf(x2, bx[jj * 4 + 2]);
          float yy2 = fminf(y2, bx[jj * 4 + 3]);
          float w = fmaxf(xx2 - xx1, 0.0f);
          float h = fmaxf(yy2 - yy1, 0.0f);
          float inter = w * h;
          float iou = inter / (ai + area[jj] - inter + 1e-9f);
          if (iou > 0.6f) supp[jj] = 1;
        }
      }
    }
    __syncthreads();
  }

  if (tid == 0) {
    int rank = 0;
    for (int i = 0; i < TOPK; ++i) {
      if (!supp[i]) {
        pos[i] = (rank < KEEP) ? rank : -1;
        ++rank;
      } else {
        pos[i] = -1;
      }
    }
    srank = rank < KEEP ? rank : KEEP;
  }
  __syncthreads();

  float* op = outp + (size_t)b * KEEP * 4;
  for (int i = tid; i < TOPK; i += nt) {
    int r = pos[i];
    if (r >= 0) {
      op[r * 4 + 0] = bx[i * 4 + 0];
      op[r * 4 + 1] = bx[i * 4 + 1];
      op[r * 4 + 2] = bx[i * 4 + 2];
      op[r * 4 + 3] = bx[i * 4 + 3];
    }
  }
  const int r0 = srank;
  for (int i = r0 + tid; i < KEEP; i += nt) {
    op[i * 4 + 0] = 0.0f; op[i * 4 + 1] = 0.0f;
    op[i * 4 + 2] = 0.0f; op[i * 4 + 3] = 0.0f;
  }
}

// ---------------------------------------------------------------------------
extern "C" void kernel_launch(void* const* d_in, const int* in_sizes, int n_in,
                              void* d_out, int out_size, void* d_ws, size_t ws_size,
                              hipStream_t stream) {
  (void)in_sizes; (void)n_in; (void)out_size; (void)ws_size;
  const float* features = (const float*)d_in[0];
  const float* conv1_w  = (const float*)d_in[1];
  const float* conv1_b  = (const float*)d_in[2];
  const float* cls_w    = (const float*)d_in[3];
  const float* cls_b    = (const float*)d_in[4];
  const float* reg_w    = (const float*)d_in[5];
  const float* reg_b    = (const float*)d_in[6];
  const int*   image_h  = (const int*)d_in[7];
  const int*   image_w  = (const int*)d_in[8];

  float* out = (float*)d_out;
  char* ws = (char*)d_ws;
  __bf16* B1    = (__bf16*)(ws + OFF_B1);
  __bf16* Bcls  = (__bf16*)(ws + OFF_BCLS);
  __bf16* Breg  = (__bf16*)(ws + OFF_BREG);
  __bf16* xin   = (__bf16*)(ws + OFF_XIN);
  __bf16* xmid  = (__bf16*)(ws + OFF_XMID);
  float*  boxes = (float*)(ws + OFF_BOXES);
  float*  sscr  = (float*)(ws + OFF_SSC);
  float*  topb  = (float*)(ws + OFF_TOPB);

  // activation repack: zero-fill padded NHWC bf16, then transpose-convert
  fill_xin_kernel<<<(XIN_WORDS + 255) / 256, 256, 0, stream>>>((unsigned*)xin);
  pack_feat_kernel<<<BB * 50 * 3 * 64, 256, 0, stream>>>(features, xin);

  // weight packing (fp32 -> bf16, GEMM-friendly layouts)
  pack3_kernel<<<(MIDc * K3 + 255) / 256, 256, 0, stream>>>(conv1_w, B1);
  pack1x1_kernel<<<(16 * MIDc + 255) / 256, 256, 0, stream>>>(cls_w, Bcls, Acnt, 16);
  pack1x1_kernel<<<(64 * MIDc + 255) / 256, 256, 0, stream>>>(reg_w, Breg, 60, 64);

  // 3x3 conv + ReLU  (barrier-free WMMA implicit GEMM)
  conv3x3_wmma<<<dim3(MTOT / 16, 1), 256, 0, stream>>>(xin, B1, conv1_b, xmid);

  // 1x1 convs (WMMA GEMMs) -> directly into d_out scores/deltas regions
  conv1x1_wmma<<<dim3(MTOT / 64, 1), 128, 0, stream>>>(xmid, Bcls, cls_b,
                                                       out + OUT_SCORES, Acnt);
  conv1x1_wmma<<<dim3(MTOT / 64, 4), 128, 0, stream>>>(xmid, Breg, reg_b,
                                                       out + OUT_DELTAS, 60);

  // anchor decode + sigmoid
  decode_kernel<<<(BB * NANCH + 255) / 256, 256, 0, stream>>>(
      out + OUT_SCORES, out + OUT_DELTAS, image_h, image_w, boxes, sscr);

  // per-image top-2000 and NMS
  topk_kernel<<<BB, 256, 0, stream>>>(sscr, boxes, topb);
  nms_kernel<<<BB, 256, 0, stream>>>(topb, out + OUT_PROPS);
}
